// gnn_encoder_88734024335671
// MI455X (gfx1250) — matrix operand order repaired and checked by reading, not compiled, not run on hardware
//
#include <hip/hip_runtime.h>
#include <stdint.h>
#include <math.h>

#define N_NODES 100000
#define N_EDGES 300000
#define HIDDEN  256
#define N_TYPES 8
#define N_STEPS 5
#define H3      (3 * HIDDEN)
#define KP      (HIDDEN / 2)    // 128 bf16 K-pairs
#define NFRAG   2048            // 8 ksteps * 32 lanes * 8 dwords = one 16x256 bf16 tile

typedef __attribute__((ext_vector_type(16))) __bf16 v16bf;
typedef __attribute__((ext_vector_type(8)))  float  v8f;

union ATile { v16bf v; uint32_t u[8]; };

__device__ __forceinline__ uint32_t pack_bf2(float lo, float hi) {
  // truncating f32 -> bf16 pair packed into one dword (lo in [15:0])
  return (__float_as_uint(hi) & 0xFFFF0000u) | (__float_as_uint(lo) >> 16);
}

__device__ __forceinline__ v8f wmma_bf16(const ATile& a, const ATile& b, v8f c) {
  return __builtin_amdgcn_wmma_f32_16x16x32_bf16(false, a.v, false, b.v,
                                                 (short)0, c, false, false);
}

// Stage one 16x256 f32 tile into LDS as bf16 fragments in WMMA per-lane order.
// slot s = ((kstep*32 + lane)*8 + v):  row = lane&15, half = lane>>4,
// v0 in {0,4} -> 8 consecutive floats starting at kstep*32 + 16*half + (v0?8:0).
__device__ __forceinline__ void stage_tile(const float* rowptr, int kbase,
                                           uint32_t* slot4) {
  float4 f0 = *(const float4*)(rowptr + kbase);
  float4 f1 = *(const float4*)(rowptr + kbase + 4);
  slot4[0] = pack_bf2(f0.x, f0.y);
  slot4[1] = pack_bf2(f0.z, f0.w);
  slot4[2] = pack_bf2(f1.x, f1.y);
  slot4[3] = pack_bf2(f1.z, f1.w);
}

__device__ __forceinline__ void load_frag8(const uint32_t* p, ATile& a) {
  const uint4* q = (const uint4*)p;
  uint4 x = q[0], y = q[1];
  a.u[0] = x.x; a.u[1] = x.y; a.u[2] = x.z; a.u[3] = x.w;
  a.u[4] = y.x; a.u[5] = y.y; a.u[6] = y.z; a.u[7] = y.w;
}

// ---------------------------------------------------------------- utilities
__global__ void zero_f32(float* p, int n) {
  int i = blockIdx.x * blockDim.x + threadIdx.x;
  if (i < n) p[i] = 0.f;
}
__global__ void zero_i32(int* p, int n) {
  int i = blockIdx.x * blockDim.x + threadIdx.x;
  if (i < n) p[i] = 0;
}

__global__ void count_types(const int* et, int* cnt) {
  int e = blockIdx.x * blockDim.x + threadIdx.x;
  if (e < N_EDGES) atomicAdd(&cnt[et[e]], 1);
}

__global__ void scan_types(const int* cnt, int* off, int* cur) {
  int o = 0;
  for (int t = 0; t < N_TYPES; ++t) {
    off[t] = o;
    cur[t] = o;
    o += (cnt[t] + 15) & ~15;
  }
  off[N_TYPES] = o;
}

__global__ void fill_buckets(const int* ei, const int* et, int* cur,
                             int* sb, int* db) {
  int e = blockIdx.x * blockDim.x + threadIdx.x;
  if (e >= N_EDGES) return;
  int t = et[e];
  int p = atomicAdd(&cur[t], 1);
  sb[p] = ei[e];
  db[p] = ei[N_EDGES + e];
}

__global__ void pad_buckets(const int* cnt, const int* off, int* sb, int* db) {
  int t = blockIdx.x;
  int i = threadIdx.x;
  int pos = off[t] + cnt[t] + i;
  if (pos < off[t + 1]) { sb[pos] = 0; db[pos] = N_NODES; }
}

__global__ void embed_kernel(const int* x, const float* emb, float* h) {
  int i = blockIdx.x * blockDim.x + threadIdx.x;
  if (i >= N_NODES * HIDDEN) return;
  int nd = i / HIDDEN, c = i - nd * HIDDEN;
  h[i] = emb[(size_t)x[nd] * HIDDEN + c];
}

// W_msg[t][k][n] -> B-fragment order: Wf[t][ct][kstep][lane][v]
// col = ct*16 + (lane&15), kp = kstep*16 + 8*(lane>>4) + v
__global__ void pack_wmsg_frag(const float* W, uint32_t* P) {
  int i = blockIdx.x * blockDim.x + threadIdx.x;
  if (i >= N_TYPES * 16 * NFRAG) return;
  int v = i & 7;
  int lane = (i >> 3) & 31;
  int kstep = (i >> 8) & 7;
  int ct = (i >> 11) & 15;
  int t = i >> 15;
  int col = ct * 16 + (lane & 15);
  int kp = kstep * 16 + 8 * (lane >> 4) + v;
  const float* w = W + (size_t)t * HIDDEN * HIDDEN;
  P[i] = pack_bf2(w[(size_t)(2 * kp) * HIDDEN + col],
                  w[(size_t)(2 * kp + 1) * HIDDEN + col]);
}

// W[j][k] ([3H][H], used as x @ W^T) -> B-fragment order: Pf[ct][kstep][lane][v]
// ct in [0,48): output col j = ct*16 + (lane&15); kp = kstep*16 + 8*(lane>>4) + v
__global__ void pack_wT_frag(const float* W, uint32_t* P) {
  int i = blockIdx.x * blockDim.x + threadIdx.x;
  if (i >= 48 * NFRAG) return;
  int v = i & 7;
  int lane = (i >> 3) & 31;
  int kstep = (i >> 8) & 7;
  int ct = i >> 11;
  int j = ct * 16 + (lane & 15);
  int kp = kstep * 16 + 8 * (lane >> 4) + v;
  P[i] = pack_bf2(W[(size_t)j * HIDDEN + 2 * kp],
                  W[(size_t)j * HIDDEN + 2 * kp + 1]);
}

// ---------------------------------------------------------------- messages
// One block (16 waves / 512 threads) per 16-edge tile.
__global__ __launch_bounds__(512)
void msg_kernel(const float* __restrict__ h, const uint32_t* __restrict__ Wf,
                const float* __restrict__ bmsg, const int* __restrict__ off,
                const int* __restrict__ sb, const int* __restrict__ db,
                float* __restrict__ aggr) {
  __shared__ uint32_t sA[NFRAG];
  __shared__ int sdst[16];

  const int tid = threadIdx.x;
  const int base = blockIdx.x * 16;
  const int total = off[N_TYPES];
  if (base >= total) return;                 // block-uniform exit

  int t = 0;
  #pragma unroll
  for (int tt = 1; tt < N_TYPES; ++tt)
    if (base >= off[tt]) t = tt;

  // ---- stage 1: A tile -> LDS
  {
    const int s0 = tid * 4;
    const int kstep = s0 >> 8;
    const int flane = (s0 >> 3) & 31;
    const int v0 = s0 & 7;
    const int row = flane & 15;
    const int half = flane >> 4;
    const int kbase = kstep * 32 + 16 * half + (v0 ? 8 : 0);
    const int srcRow = sb[base + row];
    stage_tile(h + (size_t)srcRow * HIDDEN, kbase, &sA[s0]);
    if (tid < 16) sdst[tid] = db[base + tid];
  }
  __syncthreads();

  // ---- stage 2: WMMA over K, one column tile per wave
  const int wave = tid >> 5;
  const int lane = tid & 31;
  const int row = lane & 15;
  const int half = lane >> 4;
  const int col = wave * 16 + row;
  const uint32_t* bbase = Wf + ((size_t)(t * 16 + wave) * NFRAG) + lane * 8;

  v8f acc = {0.f, 0.f, 0.f, 0.f, 0.f, 0.f, 0.f, 0.f};
  #pragma unroll
  for (int kstep = 0; kstep < 8; ++kstep) {
    ATile a, b;
    load_frag8(&sA[(kstep * 32 + lane) * 8], a);
    load_frag8(bbase + kstep * 256, b);
    acc = wmma_bf16(a, b, acc);
  }

  const float bias = bmsg[t * HIDDEN + col];
  #pragma unroll
  for (int i = 0; i < 8; ++i) {
    int m = i + 8 * half;
    int d = sdst[m];
    atomicAdd(&aggr[(size_t)d * HIDDEN + col], acc[i] + bias);
  }
}

// ---------------------------------------------------------------- GRU cell
// One block (16 waves) per 16-node tile; wave w owns hidden cols [16w,16w+16).
__global__ __launch_bounds__(512)
void gru_kernel(const float* __restrict__ aggr, const float* __restrict__ hcur,
                const uint32_t* __restrict__ Pih, const uint32_t* __restrict__ Phh,
                const float* __restrict__ bih, const float* __restrict__ bhh,
                float* __restrict__ hnext) {
  __shared__ uint32_t sAg[NFRAG];
  __shared__ uint32_t sH[NFRAG];

  const int tid = threadIdx.x;
  const int nodebase = blockIdx.x * 16;

  // ---- stage 1: both A tiles -> LDS
  {
    const int s0 = tid * 4;
    const int kstep = s0 >> 8;
    const int flane = (s0 >> 3) & 31;
    const int v0 = s0 & 7;
    const int row = flane & 15;
    const int half = flane >> 4;
    const int kbase = kstep * 32 + 16 * half + (v0 ? 8 : 0);
    stage_tile(aggr + (size_t)(nodebase + row) * HIDDEN, kbase, &sAg[s0]);
    stage_tile(hcur + (size_t)(nodebase + row) * HIDDEN, kbase, &sH[s0]);
  }
  __syncthreads();

  // ---- stage 2
  const int wave = tid >> 5;
  const int lane = tid & 31;
  const int row = lane & 15;
  const int half = lane >> 4;
  const int j = wave * 16 + row;

  // gate g lives at column tile (wave + 16*g) of the 48-tile packed matrices
  const uint32_t* pihR = Pih + ((size_t)(wave)      * NFRAG) + lane * 8;
  const uint32_t* pihZ = Pih + ((size_t)(wave + 16) * NFRAG) + lane * 8;
  const uint32_t* pihN = Pih + ((size_t)(wave + 32) * NFRAG) + lane * 8;
  const uint32_t* phhR = Phh + ((size_t)(wave)      * NFRAG) + lane * 8;
  const uint32_t* phhZ = Phh + ((size_t)(wave + 16) * NFRAG) + lane * 8;
  const uint32_t* phhN = Phh + ((size_t)(wave + 32) * NFRAG) + lane * 8;

  v8f cIR = {0.f,0.f,0.f,0.f,0.f,0.f,0.f,0.f};
  v8f cIZ = cIR, cIN = cIR, cHR = cIR, cHZ = cIR, cHN = cIR;

  for (int kstep = 0; kstep < 8; ++kstep) {
    ATile aA, aH, b;
    load_frag8(&sAg[(kstep * 32 + lane) * 8], aA);
    load_frag8(&sH[(kstep * 32 + lane) * 8], aH);
    const int ko = kstep * 256;

    load_frag8(pihR + ko, b); cIR = wmma_bf16(aA, b, cIR);
    load_frag8(pihZ + ko, b); cIZ = wmma_bf16(aA, b, cIZ);
    load_frag8(pihN + ko, b); cIN = wmma_bf16(aA, b, cIN);
    load_frag8(phhR + ko, b); cHR = wmma_bf16(aH, b, cHR);
    load_frag8(phhZ + ko, b); cHZ = wmma_bf16(aH, b, cHZ);
    load_frag8(phhN + ko, b); cHN = wmma_bf16(aH, b, cHN);
  }

  #pragma unroll
  for (int i = 0; i < 8; ++i) {
    int m = i + 8 * half;
    float ir = cIR[i] + bih[j];
    float hrv = cHR[i] + bhh[j];
    float iz = cIZ[i] + bih[HIDDEN + j];
    float hz = cHZ[i] + bhh[HIDDEN + j];
    float in_ = cIN[i] + bih[2 * HIDDEN + j];
    float hn = cHN[i] + bhh[2 * HIDDEN + j];
    float r = 1.f / (1.f + __expf(-(ir + hrv)));
    float z = 1.f / (1.f + __expf(-(iz + hz)));
    float nn = tanhf(in_ + r * hn);
    float hp = hcur[(size_t)(nodebase + m) * HIDDEN + j];
    hnext[(size_t)(nodebase + m) * HIDDEN + j] = (1.f - z) * nn + z * hp;
  }
}

// ---------------------------------------------------------------- launch
extern "C" void kernel_launch(void* const* d_in, const int* in_sizes, int n_in,
                              void* d_out, int out_size, void* d_ws, size_t ws_size,
                              hipStream_t stream) {
  (void)in_sizes; (void)n_in; (void)out_size; (void)ws_size;

  const int*   x    = (const int*)d_in[0];
  const int*   ei   = (const int*)d_in[1];
  const int*   et   = (const int*)d_in[2];
  const float* emb  = (const float*)d_in[3];
  const float* Wmsg = (const float*)d_in[4];
  const float* bmsg = (const float*)d_in[5];
  const float* wih  = (const float*)d_in[6];
  const float* whh  = (const float*)d_in[7];
  const float* bih  = (const float*)d_in[8];
  const float* bhh  = (const float*)d_in[9];

  char* wsp = (char*)d_ws;
  auto carve = [&](size_t bytes) -> char* {
    char* p = wsp;
    wsp += (bytes + 255) & ~(size_t)255;
    return p;
  };

  float*    hA   = (float*)carve(sizeof(float) * (size_t)N_NODES * HIDDEN);
  float*    hB   = (float*)carve(sizeof(float) * (size_t)N_NODES * HIDDEN);
  float*    aggr = (float*)carve(sizeof(float) * (size_t)(N_NODES + 1) * HIDDEN);
  uint32_t* Wf   = (uint32_t*)carve(sizeof(uint32_t) * (size_t)N_TYPES * 16 * NFRAG);
  uint32_t* Pih  = (uint32_t*)carve(sizeof(uint32_t) * (size_t)48 * NFRAG);
  uint32_t* Phh  = (uint32_t*)carve(sizeof(uint32_t) * (size_t)48 * NFRAG);
  const int BUCK = N_EDGES + 16 * N_TYPES;
  int* sb   = (int*)carve(sizeof(int) * BUCK);
  int* db   = (int*)carve(sizeof(int) * BUCK);
  int* ints = (int*)carve(sizeof(int) * 32);
  int* cnt = ints;          // [8]
  int* off = ints + 8;      // [9]
  int* cur = ints + 17;     // [8]

  // one-time (per launch) preprocessing
  zero_i32<<<1, 32, 0, stream>>>(ints, 25);
  count_types<<<(N_EDGES + 255) / 256, 256, 0, stream>>>(et, cnt);
  scan_types<<<1, 1, 0, stream>>>(cnt, off, cur);
  fill_buckets<<<(N_EDGES + 255) / 256, 256, 0, stream>>>(ei, et, cur, sb, db);
  pad_buckets<<<N_TYPES, 16, 0, stream>>>(cnt, off, sb, db);
  embed_kernel<<<(N_NODES * HIDDEN + 255) / 256, 256, 0, stream>>>(x, emb, hA);
  pack_wmsg_frag<<<(N_TYPES * 16 * NFRAG + 255) / 256, 256, 0, stream>>>(Wmsg, Wf);
  pack_wT_frag<<<(48 * NFRAG + 255) / 256, 256, 0, stream>>>(wih, Pih);
  pack_wT_frag<<<(48 * NFRAG + 255) / 256, 256, 0, stream>>>(whh, Phh);

  const int MSG_TILES = N_EDGES / 16 + N_TYPES;  // upper bound on padded tiles
  float* hcur = hA;
  for (int s = 0; s < N_STEPS; ++s) {
    zero_f32<<<((N_NODES + 1) * HIDDEN + 255) / 256, 256, 0, stream>>>(
        aggr, (N_NODES + 1) * HIDDEN);
    msg_kernel<<<MSG_TILES, 512, 0, stream>>>(hcur, Wf, bmsg, off, sb, db, aggr);
    float* hnext = (s == N_STEPS - 1) ? (float*)d_out
                                      : ((hcur == hA) ? hB : hA);
    gru_kernel<<<N_NODES / 16, 512, 0, stream>>>(aggr, hcur, Pih, Phh,
                                                 bih, bhh, hnext);
    hcur = hnext;
  }
}
